// UniRNN_22771916604116
// MI455X (gfx1250) — matrix operand order repaired and checked by reading, not compiled, not run on hardware
//
#include <hip/hip_runtime.h>

// ---------------------------------------------------------------------------
// 2-layer tanh RNN for MI455X (gfx1250), wave32 + WMMA bf16.
//
// Kernel 1 (xp_gemm): x_proj[t,b,:] = a[t,b,:] @ W_ih1^T + b_ih1
//   - full-chip bf16 WMMA GEMM, M = T*B = 131072, N = 256, K = 256
//   - written into d_out's [T,B,H] region (staging; each tile is consumed by
//     the scan at step t and overwritten with h2[t] at the same (t,b,n)
//     location by the same wave -> no races, no workspace).
//
// Kernel 2 (rnn_scan): sequential scan. 16 workgroups (one per 16-row batch
//   slab; batch rows are independent in the recurrence), 16 waves each.
//   Wave w owns hidden columns [16w,16w+16) of both layers.
//   - W_hh1, W_ih2 pinned in VGPRs in WMMA B-operand layout (128 VGPRs)
//   - W_hh2 staged in LDS in B-operand per-lane layout (keeps per-wave VGPR
//     demand ~190 so nothing spills into the serial loop)
//   - h-states ping-pong in LDS as bf16, fp32 WMMA accumulation,
//     hardware v_tanh_f32 (builtin if available, else inline asm).
// ---------------------------------------------------------------------------

typedef __attribute__((ext_vector_type(16))) __bf16 v16bf;
typedef __attribute__((ext_vector_type(8)))  float  v8f;

#define T_STEPS 512
#define BATCH   256
#define HID     256
#define LDSS    264                 // LDS row stride in bf16 (256 + 8 pad)
#define WLDS_BYTES (16 * 8 * 1024)  // 16 waves x 8 k-chunks x 1KB = 128KB

union V16BF {
    v16bf v;
    uint4 q[2];
};

union PACK8 {
    uint4 q;
    __bf16 h[8];
};

__device__ __forceinline__ float fast_tanh(float x) {
#if __has_builtin(__builtin_amdgcn_tanhf)
    return __builtin_amdgcn_tanhf(x);   // v_tanh_f32 (gfx1250 transcendental)
#else
    // Guaranteed hardware path: TRANS op + required 1-op separation
    // (ISA: 1 independent op or V_NOP after a TRANS before use of result).
    float r;
    asm("v_tanh_f32 %0, %1\n\tv_nop" : "=v"(r) : "v"(x));
    return r;
#endif
}

__device__ __forceinline__ v8f wmma_bf16(v16bf a, v16bf b, v8f c) {
    // D = A(16x32 bf16) * B(32x16 bf16) + C(16x16 f32)
    return __builtin_amdgcn_wmma_f32_16x16x32_bf16(
        /*neg_a=*/false, a, /*neg_b=*/false, b,
        /*c_mod=*/(short)0, c, /*reuse_a=*/false, /*reuse_b=*/false);
}

// B-operand chunk: 16 contiguous fp32 -> bf16 (native packed converts).
// Layout: lane L holds column n = L%16; lanes 0-15 K = kc*32+0..15,
// lanes 16-31 K = kc*32+16..31; element e <-> K = kbase + e (contiguous).
__device__ __forceinline__ v16bf ldB_g(const float* __restrict__ p) {
    v16bf r;
#pragma unroll
    for (int j = 0; j < 4; ++j) {
        float4 v = ((const float4*)p)[j];
        r[j * 4 + 0] = (__bf16)v.x;
        r[j * 4 + 1] = (__bf16)v.y;
        r[j * 4 + 2] = (__bf16)v.z;
        r[j * 4 + 3] = (__bf16)v.w;
    }
    return r;
}

// A-operand chunk from a global fp32 row (row = this lane's M row).
// Elements 0-15 <-> K = k0..k0+15 (contiguous, matching B),
// where k0 = kc*32 + (lane/16)*16.
__device__ __forceinline__ v16bf ldA_g(const float* __restrict__ row, int k0) {
    v16bf r;
#pragma unroll
    for (int h = 0; h < 2; ++h) {
        const float4* s = (const float4*)(row + k0 + h * 8);
#pragma unroll
        for (int j = 0; j < 2; ++j) {
            float4 v = s[j];
            const int e = h * 8 + j * 4;
            r[e + 0] = (__bf16)v.x;
            r[e + 1] = (__bf16)v.y;
            r[e + 2] = (__bf16)v.z;
            r[e + 3] = (__bf16)v.w;
        }
    }
    return r;
}

// A-operand chunk from a bf16 LDS state buffer (two 16B ds_load_b128).
__device__ __forceinline__ v16bf ldA_lds(const __bf16* buf, int kc,
                                         int lrow, int lhi) {
    V16BF r;
    const __bf16* p = buf + lrow * LDSS + kc * 32 + lhi * 16;
    r.q[0] = *(const uint4*)(p);
    r.q[1] = *(const uint4*)(p + 8);
    return r.v;
}

// W_hh2 LDS blob addressing: region (wave,kc) is 64 uint4; lane's two 16B
// halves XOR-swizzled by one slot to spread LDS banks.
__device__ __forceinline__ int wblob_idx0(int lane) {
    return (lane * 2) ^ ((lane >> 3) & 1);
}

__device__ __forceinline__ v16bf ldW_lds(const uint4* wbuf, int wave, int kc,
                                         int lane) {
    V16BF r;
    const int base = (wave * 8 + kc) * 64;
    const int i0 = wblob_idx0(lane);
    r.q[0] = wbuf[base + i0];
    r.q[1] = wbuf[base + (i0 ^ 1)];
    return r.v;
}

// Convert 8 contiguous fp32 -> 8 bf16 and store 16B into LDS.
__device__ __forceinline__ void cvt8_lds(__bf16* dst,
                                         const float* __restrict__ src) {
    float4 a = ((const float4*)src)[0];
    float4 b = ((const float4*)src)[1];
    PACK8 p;
    p.h[0] = (__bf16)a.x; p.h[1] = (__bf16)a.y;
    p.h[2] = (__bf16)a.z; p.h[3] = (__bf16)a.w;
    p.h[4] = (__bf16)b.x; p.h[5] = (__bf16)b.y;
    p.h[6] = (__bf16)b.z; p.h[7] = (__bf16)b.w;
    *(uint4*)dst = p.q;
}

// ---------------------------------------------------------------------------
// Kernel 1: x_proj GEMM (whole chip). One 16x16 output tile per wave,
// 8 waves / block, 16384 blocks -> 131072 tiles.
// ---------------------------------------------------------------------------
__global__ __launch_bounds__(256) void xp_gemm(
    const float* __restrict__ a,       // [T*B, 256]
    const float* __restrict__ W_ih1,   // [256, 256]
    const float* __restrict__ b_ih1,   // [256]
    float* __restrict__ out)           // [T*B, 256] (staging region of d_out)
{
    const int tid  = threadIdx.x;
    const int wave = tid >> 5;
    const int lane = tid & 31;
    const int lrow = lane & 15;
    const int lhi  = lane >> 4;

    const int tile = blockIdx.x * 8 + wave;   // 0 .. 131071
    const int mt = tile >> 4;                 // 0 .. 8191
    const int nt = tile & 15;                 // 0 .. 15
    const int m0 = mt * 16;
    const int n0 = nt * 16;

    const float* arow = a     + (size_t)(m0 + lrow) * HID;
    const float* wrow = W_ih1 + (size_t)(n0 + lrow) * HID;

    const float bias = b_ih1[n0 + lrow];
    v8f acc;
#pragma unroll
    for (int r = 0; r < 8; ++r) acc[r] = bias;

#pragma unroll
    for (int kc = 0; kc < 8; ++kc) {
        v16bf A = ldA_g(arow, kc * 32 + lhi * 16);
        v16bf B = ldB_g(wrow + kc * 32 + lhi * 16);
        acc = wmma_bf16(A, B, acc);
    }

    // C layout: VGPR r -> m = r + lhi*8, n = lrow
    float* o = out + (size_t)(m0 + lhi * 8) * HID + n0 + lrow;
#pragma unroll
    for (int r = 0; r < 8; ++r) o[(size_t)r * HID] = acc[r];
}

// ---------------------------------------------------------------------------
// Kernel 2: sequential 2-layer scan. 16 blocks x 512 threads (16 waves).
// Dynamic LDS: 128KB W_hh2 blobs (after the static state buffers).
// ---------------------------------------------------------------------------
__global__ __launch_bounds__(512) void rnn_scan(
    const float* __restrict__ h1_0,
    const float* __restrict__ h2_0,
    const float* __restrict__ W_hh1,
    const float* __restrict__ b_hh1,
    const float* __restrict__ W_ih2,
    const float* __restrict__ W_hh2,
    const float* __restrict__ b_ih2,
    const float* __restrict__ b_hh2,
    float* __restrict__ out)           // [T*B*H] outputs ++ [B*H] h1_f ++ [B*H] h2_f
{
    __shared__ __bf16 h1buf[2][16 * LDSS];
    __shared__ __bf16 h2buf[16 * LDSS];
    extern __shared__ char dynsmem[];
    uint4* const wbuf = (uint4*)dynsmem;   // W_hh2, B-operand per-lane layout

    const int tid  = threadIdx.x;
    const int wave = tid >> 5;
    const int lane = tid & 31;
    const int lrow = lane & 15;
    const int lhi  = lane >> 4;
    const int b0   = blockIdx.x * 16;   // batch slab
    const int nb   = wave * 16;         // hidden-column block owned by this wave

    // --- W_hh1, W_ih2 persistent in VGPRs; W_hh2 staged into LDS ---
    v16bf w_hh1[8], w_ih2[8];
    {
        const float* r1 = W_hh1 + (size_t)(nb + lrow) * HID;
        const float* r2 = W_ih2 + (size_t)(nb + lrow) * HID;
        const float* r3 = W_hh2 + (size_t)(nb + lrow) * HID;
        const int i0 = wblob_idx0(lane);
#pragma unroll
        for (int kc = 0; kc < 8; ++kc) {
            const int k0 = kc * 32 + lhi * 16;
            w_hh1[kc] = ldB_g(r1 + k0);
            w_ih2[kc] = ldB_g(r2 + k0);
            V16BF u;
            u.v = ldB_g(r3 + k0);
            const int base = (wave * 8 + kc) * 64;
            wbuf[base + i0]       = u.q[0];
            wbuf[base + (i0 ^ 1)] = u.q[1];
        }
    }
    const float bias1 = b_hh1[nb + lrow];                     // b_ih1 already in xp
    const float bias2 = b_ih2[nb + lrow] + b_hh2[nb + lrow];

    // --- initial states -> LDS (bf16). 512 threads x 8 elems = 4096 ---
    {
        const int idx = tid * 8;
        const int row = idx >> 8;
        const int col = idx & 255;
        cvt8_lds(&h1buf[0][row * LDSS + col], h1_0 + (size_t)(b0 + row) * HID + col);
        cvt8_lds(&h2buf[row * LDSS + col],    h2_0 + (size_t)(b0 + row) * HID + col);
    }
    __syncthreads();

    float* const outputs = out;
    float* const h1f = out + (size_t)T_STEPS * BATCH * HID;
    float* const h2f = h1f + (size_t)BATCH * HID;

#pragma unroll 1
    for (int t = 0; t < T_STEPS; ++t) {
        const int ping = t & 1;
        const int pong = ping ^ 1;

        // ---- layer 1: acc1 = xp[t] + b_hh1 + h1_prev @ W_hh1^T ----
        const float* xp = outputs + (size_t)(t * BATCH + b0 + lhi * 8) * HID + nb + lrow;
        v8f acc1;
#pragma unroll
        for (int r = 0; r < 8; ++r) acc1[r] = xp[(size_t)r * HID] + bias1;

        if (t + 1 < T_STEPS) {
            const float* xpn = outputs + (size_t)((t + 1) * BATCH + b0 + lhi * 8) * HID + nb + lrow;
            __builtin_prefetch(xpn, 0, 0);   // global_prefetch next step's tile
        }

#pragma unroll
        for (int kc = 0; kc < 8; ++kc) {
            v16bf A = ldA_lds(&h1buf[ping][0], kc, lrow, lhi);
            acc1 = wmma_bf16(A, w_hh1[kc], acc1);
        }
#pragma unroll
        for (int r = 0; r < 8; ++r) acc1[r] = fast_tanh(acc1[r]);

        // write h1_t (bf16) into the pong buffer (pong's last readers were
        // step t-1's layer-1, already fenced by t-1's barriers)
        {
            __bf16* d = &h1buf[pong][(lhi * 8) * LDSS + nb + lrow];
#pragma unroll
            for (int r = 0; r < 8; ++r) d[r * LDSS] = (__bf16)acc1[r];
        }
        if (t == T_STEPS - 1) {
            float* d = h1f + (size_t)(b0 + lhi * 8) * HID + nb + lrow;
#pragma unroll
            for (int r = 0; r < 8; ++r) d[(size_t)r * HID] = acc1[r];
        }
        __syncthreads();   // S1: h1_t visible to all waves

        // ---- layer 2: acc2 = b_ih2 + b_hh2 + h1_t @ W_ih2^T + h2_prev @ W_hh2^T ----
        v8f acc2;
#pragma unroll
        for (int r = 0; r < 8; ++r) acc2[r] = bias2;
#pragma unroll
        for (int kc = 0; kc < 8; ++kc) {
            v16bf A = ldA_lds(&h1buf[pong][0], kc, lrow, lhi);
            acc2 = wmma_bf16(A, w_ih2[kc], acc2);
        }
#pragma unroll
        for (int kc = 0; kc < 8; ++kc) {
            v16bf A = ldA_lds(&h2buf[0], kc, lrow, lhi);
            v16bf W = ldW_lds(wbuf, wave, kc, lane);
            acc2 = wmma_bf16(A, W, acc2);
        }
#pragma unroll
        for (int r = 0; r < 8; ++r) acc2[r] = fast_tanh(acc2[r]);
        __syncthreads();   // S2: all reads of h2_prev complete

        // write h2_t: bf16 to LDS, fp32 to out[t] (same tile this wave read xp from)
        {
            __bf16* d = &h2buf[(lhi * 8) * LDSS + nb + lrow];
#pragma unroll
            for (int r = 0; r < 8; ++r) d[r * LDSS] = (__bf16)acc2[r];
        }
        float* o = outputs + (size_t)(t * BATCH + b0 + lhi * 8) * HID + nb + lrow;
#pragma unroll
        for (int r = 0; r < 8; ++r) o[(size_t)r * HID] = acc2[r];

        if (t == T_STEPS - 1) {
            float* d = h2f + (size_t)(b0 + lhi * 8) * HID + nb + lrow;
#pragma unroll
            for (int r = 0; r < 8; ++r) d[(size_t)r * HID] = acc2[r];
        }
        // next iteration's first barrier (S1) fences h2buf writes before reads
    }
}

// ---------------------------------------------------------------------------
extern "C" void kernel_launch(void* const* d_in, const int* in_sizes, int n_in,
                              void* d_out, int out_size, void* d_ws, size_t ws_size,
                              hipStream_t stream) {
    (void)in_sizes; (void)n_in; (void)out_size; (void)d_ws; (void)ws_size;

    const float* a     = (const float*)d_in[0];
    const float* h1_0  = (const float*)d_in[1];
    const float* h2_0  = (const float*)d_in[2];
    const float* W_ih1 = (const float*)d_in[3];
    const float* W_hh1 = (const float*)d_in[4];
    const float* b_ih1 = (const float*)d_in[5];
    const float* b_hh1 = (const float*)d_in[6];
    const float* W_ih2 = (const float*)d_in[7];
    const float* W_hh2 = (const float*)d_in[8];
    const float* b_ih2 = (const float*)d_in[9];
    const float* b_hh2 = (const float*)d_in[10];
    float* out = (float*)d_out;

    // Allow the 128KB dynamic-LDS allocation (no-op if already permitted).
    (void)hipFuncSetAttribute((const void*)rnn_scan,
                              hipFuncAttributeMaxDynamicSharedMemorySize,
                              WLDS_BYTES);

    // Stage 1: x_proj into d_out's outputs region (consumed+overwritten in stage 2).
    xp_gemm<<<16384, 256, 0, stream>>>(a, W_ih1, b_ih1, out);

    // Stage 2: sequential scan, one workgroup per 16-row batch slab.
    rnn_scan<<<16, 512, WLDS_BYTES, stream>>>(h1_0, h2_0, W_hh1, b_hh1,
                                              W_ih2, W_hh2, b_ih2, b_hh2, out);
}
